// JetGAT_86775519248876
// MI455X (gfx1250) — matrix operand order repaired and checked by reading, not compiled, not run on hardware
//
#include <hip/hip_runtime.h>
#include <hip/hip_bf16.h>
#include <math.h>

#define N_NODES   50000
#define N_EDGES   400000
#define N_GRAPHS  512
#define IN_CH     6
#define POOLC     64    // final node feature dim

typedef _Float16 half8  __attribute__((ext_vector_type(8)));
typedef _Float16 half16 __attribute__((ext_vector_type(16)));
typedef float    float8 __attribute__((ext_vector_type(8)));

// ---- order-preserving float <-> uint encoding for atomic max on floats ----
__device__ __forceinline__ unsigned fenc(float f) {
  unsigned u = __float_as_uint(f);
  return (u >> 31) ? ~u : (u | 0x80000000u);
}
__device__ __forceinline__ float fdec(unsigned u) {
  unsigned b = (u >> 31) ? (u & 0x7FFFFFFFu) : ~u;
  return __uint_as_float(b);
}
__device__ __forceinline__ float gelu_exact(float x) {
  return 0.5f * x * (1.0f + erff(x * 0.70710678118654752f));
}

// ---------------------------------------------------------------------------
// K0: in_proj  h = gelu(x @ W[6,256] + b); also emit fp16 shadow
// ---------------------------------------------------------------------------
__global__ void k_in_proj(const float* __restrict__ x, const float* __restrict__ W,
                          const float* __restrict__ b, float* __restrict__ h,
                          _Float16* __restrict__ hh) {
  int n = blockIdx.x, c = threadIdx.x;                 // 256 threads
  const float* xp = x + (size_t)n * IN_CH;
  float acc = b[c];
#pragma unroll
  for (int k = 0; k < IN_CH; ++k) acc += xp[k] * W[k * 256 + c];
  float g = gelu_exact(acc);
  h[(size_t)n * 256 + c]  = g;
  hh[(size_t)n * 256 + c] = (_Float16)g;
}

// ---------------------------------------------------------------------------
// K1a: pack W[K,N] fp32 (row-major) into fp16 WMMA B-operand lane layout:
//   Wp[(((nt*(K/32)) + kb32)*32 + lane)*16 + i] = W[(kb32*32 + koff + i)*N + nt*16 + lane%16]
//   koff = (lane<16) ? 0 : 16
// so the GEMM reads B as two contiguous b128 loads per lane per K-step.
// ---------------------------------------------------------------------------
__global__ void k_pack_w(const float* __restrict__ W, _Float16* __restrict__ Wp,
                         int K, int N) {
  int idx = blockIdx.x * blockDim.x + threadIdx.x;
  if (idx >= K * N) return;
  int i    = idx & 15;
  int lane = (idx >> 4) & 31;
  int kb32 = (idx >> 9) % (K >> 5);
  int nt   = (idx >> 9) / (K >> 5);
  int koff = (lane < 16) ? 0 : 16;
  int ncol = nt * 16 + (lane & 15);
  int krow = kb32 * 32 + koff + i;
  Wp[idx] = (_Float16)W[(size_t)krow * N + ncol];
}

// ---------------------------------------------------------------------------
// K1b: dual WMMA GEMM.  Each wave computes a 16(M) x 64(N) strip of BOTH
//   Cl = A @ Wl + bl   and   Cr = A @ Wr + br
// One A-fragment load feeds 8 v_wmma_f32_16x16x32_f16 per K-step.
// A: fp16 [M,K] row-major; Wpl/Wpr: packed fp16 B layout (see k_pack_w).
// ---------------------------------------------------------------------------
__global__ void k_gemm_dual_wmma(const _Float16* __restrict__ A,
                                 const _Float16* __restrict__ Wpl,
                                 const _Float16* __restrict__ Wpr,
                                 const float* __restrict__ bl,
                                 const float* __restrict__ br,
                                 float* __restrict__ Cl, float* __restrict__ Cr,
                                 int M, int K, int Nout) {
  const int lane  = threadIdx.x & 31;
  const int wave  = threadIdx.x >> 5;
  const int nq    = Nout >> 6;                          // 64-wide N strips
  int strip = blockIdx.x * (blockDim.x >> 5) + wave;    // uniform per wave
  int total = (M >> 4) * nq;
  if (strip >= total) return;                           // whole-wave uniform exit
  const int mt  = strip / nq;
  const int ntq = strip % nq;
  const int l16   = lane & 15;
  const int row   = (mt << 4) + l16;
  const int koffA = (lane < 16) ? 0 : 16;
  const int kb32s = K >> 5;

  float8 accl0 = {}, accl1 = {}, accl2 = {}, accl3 = {};
  float8 accr0 = {}, accr1 = {}, accr2 = {}, accr3 = {};
  const _Float16* arow = A + (size_t)row * K;

  for (int kb32 = 0; kb32 < kb32s; ++kb32) {
    const int kb = kb32 << 5;
    half8 a0 = *(const half8*)(arow + kb + koffA);
    half8 a1 = *(const half8*)(arow + kb + koffA + 8);
    half16 a;
#pragma unroll
    for (int i = 0; i < 8; ++i) { a[i] = a0[i]; a[8 + i] = a1[i]; }

#pragma unroll
    for (int q = 0; q < 4; ++q) {
      const size_t boff = (((size_t)(ntq * 4 + q) * kb32s + kb32) * 32 + lane) * 16;
      half8 bL0 = *(const half8*)(Wpl + boff);
      half8 bL1 = *(const half8*)(Wpl + boff + 8);
      half8 bR0 = *(const half8*)(Wpr + boff);
      half8 bR1 = *(const half8*)(Wpr + boff + 8);
      half16 bLv, bRv;
#pragma unroll
      for (int i = 0; i < 8; ++i) {
        bLv[i] = bL0[i]; bLv[8 + i] = bL1[i];
        bRv[i] = bR0[i]; bRv[8 + i] = bR1[i];
      }
      float8 cl = (q == 0) ? accl0 : (q == 1) ? accl1 : (q == 2) ? accl2 : accl3;
      float8 cr = (q == 0) ? accr0 : (q == 1) ? accr1 : (q == 2) ? accr2 : accr3;
      cl = __builtin_amdgcn_wmma_f32_16x16x32_f16(false, a, false, bLv, (short)0, cl, false, false);
      cr = __builtin_amdgcn_wmma_f32_16x16x32_f16(false, a, false, bRv, (short)0, cr, false, false);
      if (q == 0) { accl0 = cl; accr0 = cr; }
      else if (q == 1) { accl1 = cl; accr1 = cr; }
      else if (q == 2) { accl2 = cl; accr2 = cr; }
      else { accl3 = cl; accr3 = cr; }
    }
  }

  const int rbase = (lane < 16) ? 0 : 8;
#pragma unroll
  for (int q = 0; q < 4; ++q) {
    const int ncol = (ntq * 4 + q) * 16 + l16;
    float8 cl = (q == 0) ? accl0 : (q == 1) ? accl1 : (q == 2) ? accl2 : accl3;
    float8 cr = (q == 0) ? accr0 : (q == 1) ? accr1 : (q == 2) ? accr2 : accr3;
    float bvl = bl[ncol], bvr = br[ncol];
    float* clp = Cl + (size_t)((mt << 4) + rbase) * Nout + ncol;
    float* crp = Cr + (size_t)((mt << 4) + rbase) * Nout + ncol;
#pragma unroll
    for (int r = 0; r < 8; ++r) {
      clp[(size_t)r * Nout] = cl[r] + bvl;
      crp[(size_t)r * Nout] = cr[r] + bvr;
    }
  }
}

// ---------------------------------------------------------------------------
// K2: per-(edge,head) attention score + segment max over dst (atomic).
// s = sum_d leakyrelu(xl[src]+xr[dst]+ef, 0.2) * att ; ef recomputed on the fly.
// float4-vectorized over the 64 head channels.
// ---------------------------------------------------------------------------
__global__ void k_edge_score(const float* __restrict__ xl, const float* __restrict__ xr,
                             const float* __restrict__ ea, const float* __restrict__ We,
                             const float* __restrict__ be, const float* __restrict__ att,
                             const int* __restrict__ src, const int* __restrict__ dst,
                             float* __restrict__ s, unsigned* __restrict__ smax,
                             int E, int H, int HO) {
  int t = blockIdx.x * blockDim.x + threadIdx.x;
  if (t >= E * H) return;
  int e = t / H, h = t % H;
  int si = src[e], di = dst[e];
  float4 eav = *(const float4*)(ea + (size_t)e * 4);
  const float4* xlp = (const float4*)(xl + (size_t)si * HO + h * 64);
  const float4* xrp = (const float4*)(xr + (size_t)di * HO + h * 64);
  const float4* ap  = (const float4*)(att + h * 64);
  const float4* bep = (const float4*)(be + h * 64);
  const float4* w0  = (const float4*)(We + 0 * HO + h * 64);
  const float4* w1  = (const float4*)(We + 1 * HO + h * 64);
  const float4* w2  = (const float4*)(We + 2 * HO + h * 64);
  const float4* w3  = (const float4*)(We + 3 * HO + h * 64);
  float sv = 0.0f;
#pragma unroll 4
  for (int d4 = 0; d4 < 16; ++d4) {
    float4 vl = xlp[d4], vr = xrp[d4], vb = bep[d4];
    float4 e0 = w0[d4], e1 = w1[d4], e2 = w2[d4], e3 = w3[d4];
    float4 va = ap[d4];
    float m;
    m = vl.x + vr.x + vb.x + eav.x*e0.x + eav.y*e1.x + eav.z*e2.x + eav.w*e3.x;
    sv += (m > 0.0f ? m : 0.2f * m) * va.x;
    m = vl.y + vr.y + vb.y + eav.x*e0.y + eav.y*e1.y + eav.z*e2.y + eav.w*e3.y;
    sv += (m > 0.0f ? m : 0.2f * m) * va.y;
    m = vl.z + vr.z + vb.z + eav.x*e0.z + eav.y*e1.z + eav.z*e2.z + eav.w*e3.z;
    sv += (m > 0.0f ? m : 0.2f * m) * va.z;
    m = vl.w + vr.w + vb.w + eav.x*e0.w + eav.y*e1.w + eav.z*e2.w + eav.w*e3.w;
    sv += (m > 0.0f ? m : 0.2f * m) * va.w;
  }
  s[t] = sv;
  atomicMax(smax + (size_t)di * H + h, fenc(sv));
}

// ---------------------------------------------------------------------------
// K3: w = exp(s - smax[dst]); wsum[dst] += w  (segment softmax denominator)
// ---------------------------------------------------------------------------
__global__ void k_edge_softmax(const int* __restrict__ dst, const unsigned* __restrict__ smax,
                               float* __restrict__ sw, float* __restrict__ wsum, int E, int H) {
  int t = blockIdx.x * blockDim.x + threadIdx.x;
  if (t >= E * H) return;
  int e = t / H, h = t % H;
  int di = dst[e];
  float wv = __expf(sw[t] - fdec(smax[(size_t)di * H + h]));
  sw[t] = wv;
  atomicAdd(wsum + (size_t)di * H + h, wv);
}

// ---------------------------------------------------------------------------
// K4: out[dst] += xl[src] * alpha   (one block per edge, one thread per chan)
// ---------------------------------------------------------------------------
__global__ void k_edge_aggregate(const int* __restrict__ src, const int* __restrict__ dst,
                                 const float* __restrict__ xl, const float* __restrict__ w,
                                 const float* __restrict__ wsum, float* __restrict__ out,
                                 int H, int HO) {
  int e = blockIdx.x, c = threadIdx.x;
  int h = c >> 6;
  int si = src[e], di = dst[e];
  float alpha = w[(size_t)e * H + h] / (wsum[(size_t)di * H + h] + 1e-16f);
  atomicAdd(out + (size_t)di * HO + c, xl[(size_t)si * HO + c] * alpha);
}

// ---------------------------------------------------------------------------
// K5: per-node bias + layernorm + gelu (+ residual), emit fp16 shadow
// blockDim == C (256 or 64)
// ---------------------------------------------------------------------------
__global__ void k_node_post(const float* __restrict__ out, const float* __restrict__ bias,
                            const float* __restrict__ lng, const float* __restrict__ lnb,
                            float* __restrict__ h, _Float16* __restrict__ hh,
                            int C, int residual, int write_half) {
  __shared__ float red[256];
  int n = blockIdx.x, c = threadIdx.x;
  float v = out[(size_t)n * C + c] + bias[c];
  red[c] = v; __syncthreads();
  for (int s = C >> 1; s > 0; s >>= 1) { if (c < s) red[c] += red[c + s]; __syncthreads(); }
  float mu = red[0] / (float)C; __syncthreads();
  float d = v - mu;
  red[c] = d * d; __syncthreads();
  for (int s = C >> 1; s > 0; s >>= 1) { if (c < s) red[c] += red[c + s]; __syncthreads(); }
  float var = red[0] / (float)C;
  float ln = lng[c] * d * rsqrtf(var + 1e-5f) + lnb[c];
  float g  = gelu_exact(ln);
  float hv = residual ? (h[(size_t)n * C + c] + g) : g;
  h[(size_t)n * C + c] = hv;
  if (write_half) hh[(size_t)n * C + c] = (_Float16)hv;
}

// ---------------------------------------------------------------------------
// K6: per-graph mean-sum / max pooling over nodes (64 channels)
// ---------------------------------------------------------------------------
__global__ void k_pool(const float* __restrict__ h, const int* __restrict__ batch,
                       float* __restrict__ gsum, unsigned* __restrict__ gmax,
                       float* __restrict__ gcnt) {
  int n = blockIdx.x, c = threadIdx.x;                 // 64 threads
  int g = batch[n];
  float v = h[(size_t)n * POOLC + c];
  atomicAdd(gsum + (size_t)g * POOLC + c, v);
  atomicMax(gmax + (size_t)g * POOLC + c, fenc(v));
  if (c == 0) atomicAdd(gcnt + g, 1.0f);
}

// ---------------------------------------------------------------------------
// K7: per-graph FC head: fc1(128->256)+LN+gelu, fc2(256->128)+LN+gelu, fc3(128->2)
// one block (256 threads) per graph
// ---------------------------------------------------------------------------
__global__ void k_head(const float* __restrict__ gsum, const unsigned* __restrict__ gmaxe,
                       const float* __restrict__ gcnt,
                       const float* __restrict__ W1, const float* __restrict__ b1,
                       const float* __restrict__ g1, const float* __restrict__ be1,
                       const float* __restrict__ W2, const float* __restrict__ b2,
                       const float* __restrict__ g2, const float* __restrict__ be2,
                       const float* __restrict__ W3, const float* __restrict__ b3,
                       float* __restrict__ outp) {
  __shared__ float feat[128], a1[256], a2[128], red[256];
  int g = blockIdx.x, t = threadIdx.x;
  if (t < POOLC) {
    float cnt = gcnt[g]; if (cnt < 1.0f) cnt = 1.0f;
    feat[t] = gsum[(size_t)g * POOLC + t] / cnt;
    float mx = fdec(gmaxe[(size_t)g * POOLC + t]);
    if (!(fabsf(mx) <= 3.0e38f)) mx = 0.0f;            // NaN/inf -> 0
    feat[POOLC + t] = mx;
  }
  __syncthreads();
  // fc1 (128 -> 256), LN(256), gelu
  float v = b1[t];
  for (int k = 0; k < 128; ++k) v += feat[k] * W1[k * 256 + t];
  red[t] = v; __syncthreads();
  for (int s = 128; s > 0; s >>= 1) { if (t < s) red[t] += red[t + s]; __syncthreads(); }
  float mu = red[0] * (1.0f / 256.0f); __syncthreads();
  float d = v - mu; red[t] = d * d; __syncthreads();
  for (int s = 128; s > 0; s >>= 1) { if (t < s) red[t] += red[t + s]; __syncthreads(); }
  float var = red[0] * (1.0f / 256.0f); __syncthreads();
  a1[t] = gelu_exact(g1[t] * d * rsqrtf(var + 1e-5f) + be1[t]);
  __syncthreads();
  // fc2 (256 -> 128), LN(128), gelu
  float v2 = 0.0f;
  if (t < 128) { v2 = b2[t]; for (int k = 0; k < 256; ++k) v2 += a1[k] * W2[k * 128 + t]; }
  red[t] = (t < 128) ? v2 : 0.0f; __syncthreads();
  for (int s = 128; s > 0; s >>= 1) { if (t < s) red[t] += red[t + s]; __syncthreads(); }
  float mu2 = red[0] * (1.0f / 128.0f); __syncthreads();
  float d2 = (t < 128) ? (v2 - mu2) : 0.0f;
  red[t] = d2 * d2; __syncthreads();
  for (int s = 128; s > 0; s >>= 1) { if (t < s) red[t] += red[t + s]; __syncthreads(); }
  float var2 = red[0] * (1.0f / 128.0f); __syncthreads();
  if (t < 128) a2[t] = gelu_exact(g2[t] * d2 * rsqrtf(var2 + 1e-5f) + be2[t]);
  __syncthreads();
  if (t < 2) {
    float v3 = b3[t];
    for (int k = 0; k < 128; ++k) v3 += a2[k] * W3[k * 2 + t];
    outp[(size_t)g * 2 + t] = v3;
  }
}

// ---------------------------------------------------------------------------
extern "C" void kernel_launch(void* const* d_in, const int* in_sizes, int n_in,
                              void* d_out, int out_size, void* d_ws, size_t ws_size,
                              hipStream_t stream) {
  auto F = [&](int i) { return (const float*)d_in[i]; };

  const int *batch, *edge_index;
  const float *x, *edge_attr, *ipW, *ipb;
  const float *lW[3], *lb[3], *rW[3], *rb[3], *eW[3], *eb[3], *att[3], *bias[3], *lng[3], *lnb[3];
  const float *fc1W, *fc1b, *ln1g, *ln1b, *fc2W, *fc2b, *ln2g, *ln2b, *fc3W, *fc3b;

  if (in_sizes[0] == N_NODES) {
    // jax tree_leaves order (dict keys sorted at every level)
    batch = (const int*)d_in[0]; edge_attr = F(1); edge_index = (const int*)d_in[2];
    fc1W = F(3); fc1b = F(4); fc2W = F(5); fc2b = F(6); fc3W = F(7); fc3b = F(8);
    ipW = F(9); ipb = F(10);
    for (int l = 0; l < 3; ++l) {
      int b = 11 + 10 * l;   // att, bias, lin_e.{W,b}, lin_l.{W,b}, lin_r.{W,b}, ln_b, ln_g
      att[l] = F(b + 0); bias[l] = F(b + 1);
      eW[l] = F(b + 2); eb[l] = F(b + 3);
      lW[l] = F(b + 4); lb[l] = F(b + 5);
      rW[l] = F(b + 6); rb[l] = F(b + 7);
      lnb[l] = F(b + 8); lng[l] = F(b + 9);
    }
    ln1b = F(41); ln1g = F(42); ln2b = F(43); ln2g = F(44); x = F(45);
  } else {
    // setup_inputs() insertion order
    x = F(0); edge_index = (const int*)d_in[1]; edge_attr = F(2); batch = (const int*)d_in[3];
    ipW = F(4); ipb = F(5);
    for (int l = 0; l < 3; ++l) {
      int b = 6 + 10 * l;    // lin_l.{W,b}, lin_r.{W,b}, lin_e.{W,b}, att, bias, ln_g, ln_b
      lW[l] = F(b + 0); lb[l] = F(b + 1);
      rW[l] = F(b + 2); rb[l] = F(b + 3);
      eW[l] = F(b + 4); eb[l] = F(b + 5);
      att[l] = F(b + 6); bias[l] = F(b + 7);
      lng[l] = F(b + 8); lnb[l] = F(b + 9);
    }
    fc1W = F(36); fc1b = F(37); ln1g = F(38); ln1b = F(39);
    fc2W = F(40); fc2b = F(41); ln2g = F(42); ln2b = F(43);
    fc3W = F(44); fc3b = F(45);
  }

  const int* src = edge_index;
  const int* dst = edge_index + N_EDGES;

  // ---- workspace bump allocator (256B aligned) ----
  char* p = (char*)d_ws;
  auto alloc = [&](size_t bytes) { void* r = p; p += (bytes + 255) & ~(size_t)255; return r; };
  float*    hbuf  = (float*)   alloc((size_t)N_NODES * 256 * 4);
  _Float16* hhalf = (_Float16*)alloc((size_t)N_NODES * 256 * 2);
  float*    xl    = (float*)   alloc((size_t)N_NODES * 256 * 4);
  float*    xrout = (float*)   alloc((size_t)N_NODES * 256 * 4);  // xr, then reused as agg out
  float*    sw    = (float*)   alloc((size_t)N_EDGES * 4 * 4);
  unsigned* smax  = (unsigned*)alloc((size_t)N_NODES * 4 * 4);
  float*    wsum  = (float*)   alloc((size_t)N_NODES * 4 * 4);
  _Float16* wpl   = (_Float16*)alloc((size_t)256 * 256 * 2);      // packed lin_l weights
  _Float16* wpr   = (_Float16*)alloc((size_t)256 * 256 * 2);      // packed lin_r weights
  float*    gsum  = (float*)   alloc((size_t)N_GRAPHS * POOLC * 4);
  unsigned* gmaxe = (unsigned*)alloc((size_t)N_GRAPHS * POOLC * 4);
  float*    gcnt  = (float*)   alloc((size_t)N_GRAPHS * 4);

  // ---- in_proj + gelu ----
  k_in_proj<<<N_NODES, 256, 0, stream>>>(x, ipW, ipb, hbuf, hhalf);

  // ---- GATv2 layers ----
  for (int l = 0; l < 3; ++l) {
    const int H  = (l < 2) ? 4 : 1;
    const int HO = H * 64;
    const int K  = 256;

    // pack weights into WMMA B layout (fp16)
    k_pack_w<<<(K * HO + 255) / 256, 256, 0, stream>>>(lW[l], wpl, K, HO);
    k_pack_w<<<(K * HO + 255) / 256, 256, 0, stream>>>(rW[l], wpr, K, HO);

    // dual GEMM: xl = h@Wl+bl, xr = h@Wr+br  (16x64 strip per wave, 8 WMMA/K-step)
    const int strips = (N_NODES / 16) * (HO / 64);
    k_gemm_dual_wmma<<<(strips + 3) / 4, 128, 0, stream>>>(
        hhalf, wpl, wpr, lb[l], rb[l], xl, xrout, N_NODES, K, HO);

    hipMemsetAsync(smax, 0, (size_t)N_NODES * H * 4, stream);
    hipMemsetAsync(wsum, 0, (size_t)N_NODES * H * 4, stream);

    int eh = N_EDGES * H;
    k_edge_score<<<(eh + 255) / 256, 256, 0, stream>>>(
        xl, xrout, edge_attr, eW[l], eb[l], att[l], src, dst, sw, smax, N_EDGES, H, HO);
    k_edge_softmax<<<(eh + 255) / 256, 256, 0, stream>>>(dst, smax, sw, wsum, N_EDGES, H);

    hipMemsetAsync(xrout, 0, (size_t)N_NODES * HO * 4, stream);  // reuse xr as agg buffer
    k_edge_aggregate<<<N_EDGES, HO, 0, stream>>>(src, dst, xl, sw, wsum, xrout, H, HO);

    k_node_post<<<N_NODES, HO, 0, stream>>>(
        xrout, bias[l], lng[l], lnb[l], hbuf, hhalf, HO, /*residual=*/(l < 2), /*write_half=*/(l < 2));
  }

  // ---- graph pooling (mean + max) ----
  hipMemsetAsync(gsum, 0, (size_t)N_GRAPHS * POOLC * 4, stream);
  hipMemsetAsync(gmaxe, 0, (size_t)N_GRAPHS * POOLC * 4, stream);
  hipMemsetAsync(gcnt, 0, (size_t)N_GRAPHS * 4, stream);
  k_pool<<<N_NODES, POOLC, 0, stream>>>(hbuf, batch, gsum, gmaxe, gcnt);

  // ---- FC head ----
  k_head<<<N_GRAPHS, 256, 0, stream>>>(gsum, gmaxe, gcnt,
                                       fc1W, fc1b, ln1g, ln1b,
                                       fc2W, fc2b, ln2g, ln2b,
                                       fc3W, fc3b, (float*)d_out);
}
